// LocalAttention1_54992761258369
// MI455X (gfx1250) — matrix operand order repaired and checked by reading, not compiled, not run on hardware
//
#include <hip/hip_runtime.h>
#include <hip/hip_bf16.h>
#include <math.h>

// Problem constants (match reference)
#define BB 8
#define TT 4096
#define DD 768
#define HH 12
#define HD 64
#define INTERVAL 3
#define MM (BB * TT)   // 32768 rows

typedef _Float16 v16h __attribute__((ext_vector_type(16)));
typedef _Float16 v8h  __attribute__((ext_vector_type(8)));
typedef float    v8f  __attribute__((ext_vector_type(8)));

// ---------------------------------------------------------------------------
// 1) f32 -> f16 elementwise convert
// ---------------------------------------------------------------------------
__global__ __launch_bounds__(256) void cvt_f32_to_f16(const float* __restrict__ in,
                                                      _Float16* __restrict__ out,
                                                      long n) {
    long i = (long)blockIdx.x * 256 + threadIdx.x;
    if (i < n) out[i] = (_Float16)in[i];
}

// ---------------------------------------------------------------------------
// 2) Transpose K x N (row-major) f32 weight into N x K (row-major) f16
//    Wt[n*K + k] = W[k*N + n]
// ---------------------------------------------------------------------------
__global__ __launch_bounds__(256) void transpose_w_f16(const float* __restrict__ W,
                                                       _Float16* __restrict__ Wt,
                                                       int N, int K) {
    long i = (long)blockIdx.x * 256 + threadIdx.x;
    long total = (long)N * K;
    if (i < total) {
        int n = (int)(i / K);
        int k = (int)(i % K);
        Wt[i] = (_Float16)W[(long)k * N + n];
    }
}

// ---------------------------------------------------------------------------
// 3) WMMA GEMM: C[M x N] = A[M x K](f16) * Wt[N x K](f16, row = output col) + bias
//    Compile-time variants:
//      WRITE_F16=true  : out is f16 buffer, no residual
//      WRITE_F16=false : out is f32 buffer, ADD_RES selects +residual (f32)
//    Block: 256 threads = 8 waves, wave tile 32x32, block tile 64(M) x 128(N).
//    K-loop steps of 32 using v_wmma_f32_16x16x32_f16 (4 per iteration).
// ---------------------------------------------------------------------------
template <bool WRITE_F16, bool ADD_RES>
__global__ __launch_bounds__(256) void gemm_wmma_f16(const _Float16* __restrict__ A,
                                                     const _Float16* __restrict__ Wt,
                                                     const float* __restrict__ bias,
                                                     const float* __restrict__ residual,
                                                     _Float16* __restrict__ outH,
                                                     float* __restrict__ outF,
                                                     int M, int N, int K) {
    const int lane = threadIdx.x & 31;
    const int wave = threadIdx.x >> 5;
    const int wm = wave >> 2;        // 0..1
    const int wn = wave & 3;         // 0..3
    const int mBase = blockIdx.y * 64 + wm * 32;
    const int nBase = blockIdx.x * 128 + wn * 32;
    const int l15 = lane & 15;
    const int hiHalf = lane >> 4;            // 0 for lanes 0-15, 1 for 16-31
    const int kOffA = hiHalf ? 16 : 0;       // A-fragment K base offset (contiguous, matches B)
    const int kOffB = hiHalf ? 16 : 0;       // B-fragment K base offset (contiguous, matches A)

    const _Float16* aRow0 = A + (long)(mBase + l15) * K + kOffA;
    const _Float16* aRow1 = A + (long)(mBase + 16 + l15) * K + kOffA;
    const _Float16* bRow0 = Wt + (long)(nBase + l15) * K + kOffB;
    const _Float16* bRow1 = Wt + (long)(nBase + 16 + l15) * K + kOffB;

    v8f acc[2][2] = {};

    for (int kt = 0; kt < K; kt += 32) {
        // Prefetch next K-tile of A rows (emits global_prefetch_b8; harmless at end)
        __builtin_prefetch(aRow0 + kt + 32, 0, 1);
        __builtin_prefetch(aRow1 + kt + 32, 0, 1);

        v16h afrag[2], bfrag[2];
        afrag[0] = *(const v16h*)(aRow0 + kt);
        afrag[1] = *(const v16h*)(aRow1 + kt);
        bfrag[0] = *(const v16h*)(bRow0 + kt);
        bfrag[1] = *(const v16h*)(bRow1 + kt);

#pragma unroll
        for (int mi = 0; mi < 2; ++mi)
#pragma unroll
            for (int ni = 0; ni < 2; ++ni)
                acc[mi][ni] = __builtin_amdgcn_wmma_f32_16x16x32_f16(
                    false, afrag[mi], false, bfrag[ni],
                    (short)0, acc[mi][ni], false, false);
    }

    // Epilogue: C/D layout — VGPR r, lanes 0-15: M=r, N=lane; lanes 16-31: M=8+r, N=lane-16
    const int rowHi = hiHalf ? 8 : 0;
#pragma unroll
    for (int mi = 0; mi < 2; ++mi) {
#pragma unroll
        for (int ni = 0; ni < 2; ++ni) {
            const int n = nBase + ni * 16 + l15;
            const float bn = bias[n];
            const long base = (long)(mBase + mi * 16 + rowHi) * N + n;
#pragma unroll
            for (int r = 0; r < 8; ++r) {
                const long idx = base + (long)r * N;
                float v = acc[mi][ni][r] + bn;
                if (ADD_RES) v += residual[idx];
                if (WRITE_F16) outH[idx] = (_Float16)v;
                else           outF[idx] = v;
            }
        }
    }
}

// ---------------------------------------------------------------------------
// 4) Local attention: one wave32 per (b, t, h). Each lane owns 2 of 64 dims.
//    Neighbors {t-3, t, t+3}; wave reduction via shfl_xor; masked softmax.
// ---------------------------------------------------------------------------
__global__ __launch_bounds__(256) void local_attn(const _Float16* __restrict__ qh,
                                                  const _Float16* __restrict__ kh,
                                                  const _Float16* __restrict__ vh,
                                                  _Float16* __restrict__ ah) {
    const int lane = threadIdx.x & 31;
    const int wv = blockIdx.x * 8 + (threadIdx.x >> 5);  // 0 .. B*T*H-1
    const int h = wv % HH;
    const int t = (wv / HH) % TT;
    const int b = wv / (HH * TT);

    const long qbase = ((long)(b * TT + t) * DD) + h * HD + lane * 2;
    const float qx = (float)qh[qbase];
    const float qy = (float)qh[qbase + 1];

    const int offs[3] = {-INTERVAL, 0, INTERVAL};
    float score[3];
    bool valid[3];
#pragma unroll
    for (int j = 0; j < 3; ++j) {
        const int tn = t + offs[j];
        valid[j] = (tn >= 0) && (tn < TT);
        float s = -INFINITY;
        if (valid[j]) {  // wave-uniform branch
            const long kb = ((long)(b * TT + tn) * DD) + h * HD + lane * 2;
            float p = qx * (float)kh[kb] + qy * (float)kh[kb + 1];
#pragma unroll
            for (int o = 16; o >= 1; o >>= 1) p += __shfl_xor(p, o, 32);
            s = p * 0.125f;  // 1/sqrt(64)
        }
        score[j] = s;
    }

    const float mx = fmaxf(score[1], fmaxf(score[0], score[2]));
    float e[3];
    float sum = 0.f;
#pragma unroll
    for (int j = 0; j < 3; ++j) {
        e[j] = valid[j] ? __expf(score[j] - mx) : 0.f;
        sum += e[j];
    }
    const float inv = 1.f / sum;

    float ox = 0.f, oy = 0.f;
#pragma unroll
    for (int j = 0; j < 3; ++j) {
        if (valid[j]) {  // wave-uniform
            const long vb = ((long)(b * TT + t + offs[j]) * DD) + h * HD + lane * 2;
            const float w = e[j] * inv;
            ox += w * (float)vh[vb];
            oy += w * (float)vh[vb + 1];
        }
    }
    ah[qbase]     = (_Float16)ox;
    ah[qbase + 1] = (_Float16)oy;
}

// ---------------------------------------------------------------------------
// 5) In-place LayerNorm over D=768: one 256-thread block per row (3 elems/thread)
// ---------------------------------------------------------------------------
__global__ __launch_bounds__(256) void layernorm_inplace(float* __restrict__ y,
                                                         const float* __restrict__ gamma,
                                                         const float* __restrict__ beta) {
    const int tid = threadIdx.x;
    float* p = y + (long)blockIdx.x * DD;

    float v[3], s = 0.f, s2 = 0.f;
#pragma unroll
    for (int i = 0; i < 3; ++i) {
        v[i] = p[tid + i * 256];
        s += v[i];
        s2 += v[i] * v[i];
    }

    __shared__ float sh[256];
    sh[tid] = s;
    __syncthreads();
    for (int o = 128; o > 0; o >>= 1) {
        if (tid < o) sh[tid] += sh[tid + o];
        __syncthreads();
    }
    const float sumAll = sh[0];
    __syncthreads();
    sh[tid] = s2;
    __syncthreads();
    for (int o = 128; o > 0; o >>= 1) {
        if (tid < o) sh[tid] += sh[tid + o];
        __syncthreads();
    }
    const float sqAll = sh[0];

    const float mu = sumAll * (1.0f / DD);
    const float var = sqAll * (1.0f / DD) - mu * mu;
    const float inv = rsqrtf(var + 1e-5f);
#pragma unroll
    for (int i = 0; i < 3; ++i) {
        const int c = tid + i * 256;
        p[c] = (v[i] - mu) * inv * gamma[c] + beta[c];
    }
}

// ---------------------------------------------------------------------------
// Host-side orchestration
// ---------------------------------------------------------------------------
extern "C" void kernel_launch(void* const* d_in, const int* in_sizes, int n_in,
                              void* d_out, int out_size, void* d_ws, size_t ws_size,
                              hipStream_t stream) {
    const float* x     = (const float*)d_in[0];
    const float* Wq    = (const float*)d_in[1];
    const float* bq    = (const float*)d_in[2];
    const float* Wk    = (const float*)d_in[3];
    const float* bk    = (const float*)d_in[4];
    const float* Wv    = (const float*)d_in[5];
    const float* bv    = (const float*)d_in[6];
    const float* Wo    = (const float*)d_in[7];
    const float* bo    = (const float*)d_in[8];
    const float* gamma = (const float*)d_in[9];
    const float* beta  = (const float*)d_in[10];
    float* out = (float*)d_out;

    const long NX = (long)MM * DD;          // 25,165,824 elements
    const long NW = (long)DD * DD;          // 589,824 elements

    char* ws = (char*)d_ws;
    size_t off = 0;
    _Float16* xh  = (_Float16*)(ws + off); off += (size_t)NX * 2;   // 50.3 MB
    _Float16* Wqh = (_Float16*)(ws + off); off += (size_t)NW * 2;
    _Float16* Wkh = (_Float16*)(ws + off); off += (size_t)NW * 2;
    _Float16* Wvh = (_Float16*)(ws + off); off += (size_t)NW * 2;
    _Float16* Woh = (_Float16*)(ws + off); off += (size_t)NW * 2;
    _Float16* qh  = (_Float16*)(ws + off); off += (size_t)NX * 2;
    _Float16* kh  = (_Float16*)(ws + off); off += (size_t)NX * 2;
    _Float16* vh  = (_Float16*)(ws + off); off += (size_t)NX * 2;
    _Float16* ah  = (_Float16*)(ws + off); off += (size_t)NX * 2;   // total ~257 MB

    // Stage activations and weights to f16 (weights transposed to N x K)
    cvt_f32_to_f16<<<(unsigned)((NX + 255) / 256), 256, 0, stream>>>(x, xh, NX);
    const unsigned wblk = (unsigned)((NW + 255) / 256);
    transpose_w_f16<<<wblk, 256, 0, stream>>>(Wq, Wqh, DD, DD);
    transpose_w_f16<<<wblk, 256, 0, stream>>>(Wk, Wkh, DD, DD);
    transpose_w_f16<<<wblk, 256, 0, stream>>>(Wv, Wvh, DD, DD);
    transpose_w_f16<<<wblk, 256, 0, stream>>>(Wo, Woh, DD, DD);

    // Q/K/V projections: grid tiles N(768)/128 x M(32768)/64
    dim3 gGemm(DD / 128, MM / 64);
    gemm_wmma_f16<true, false><<<gGemm, 256, 0, stream>>>(xh, Wqh, bq, nullptr, qh, nullptr, MM, DD, DD);
    gemm_wmma_f16<true, false><<<gGemm, 256, 0, stream>>>(xh, Wkh, bk, nullptr, kh, nullptr, MM, DD, DD);
    gemm_wmma_f16<true, false><<<gGemm, 256, 0, stream>>>(xh, Wvh, bv, nullptr, vh, nullptr, MM, DD, DD);

    // Local attention: one wave per (b,t,h); 8 waves per block
    const unsigned nWaves = (unsigned)(BB * TT * HH);
    local_attn<<<nWaves / 8, 256, 0, stream>>>(qh, kh, vh, ah);

    // Output projection + bias + residual -> d_out (f32)
    gemm_wmma_f16<false, true><<<gGemm, 256, 0, stream>>>(ah, Woh, bo, x, nullptr, out, MM, DD, DD);

    // LayerNorm in place on d_out
    layernorm_inplace<<<MM, 256, 0, stream>>>(out, gamma, beta);

    (void)in_sizes; (void)n_in; (void)out_size; (void)ws_size;
}
